// FlashMoeBlockWrapper_85023172592623
// MI455X (gfx1250) — compile-verified, matrix operand order checked by reading, not run on hardware
//
#include <hip/hip_runtime.h>
#include <hip/hip_bf16.h>
#include <stdint.h>

// Problem constants (from reference)
#define TT   2048
#define HH   2048
#define EE   16
#define II   768
#define TOPK 2
#define CAPPAD (TT*TOPK + EE*64)   // 5120 padded assignment slots

typedef __attribute__((ext_vector_type(16))) __bf16        v16bf;
typedef __attribute__((ext_vector_type(2)))  __bf16        v2bf;
typedef __attribute__((ext_vector_type(8)))  float         v8f;
typedef __attribute__((ext_vector_type(8)))  unsigned int  v8u;
typedef int tv4i __attribute__((vector_size(16)));         // v4i32 for async-LDS builtin

// Optional CDNA5 fast paths (degrade gracefully if builtin absent)
#if __has_builtin(__builtin_amdgcn_cvt_pk_bf16_f32)
#define USE_CVT_PK 1
#endif
#if __has_builtin(__builtin_amdgcn_global_load_async_to_lds_b128) && \
    __has_builtin(__builtin_amdgcn_s_wait_asynccnt)
#define USE_ASYNC_LDS 1
#endif

// ---------------- workspace layout (bytes) ----------------
#define OFF_COUNTS   ((size_t)0)                         // 16 int
#define OFF_OFFS     ((size_t)256)                       // 17 int
#define OFF_TOKEXP   ((size_t)512)                       // 2T int  = 16KB
#define OFF_TOKW     (OFF_TOKEXP + 16384)                // 2T f32  = 16KB
#define OFF_SLOTTOK  (OFF_TOKW   + 16384)                // CAPPAD int = 20KB
#define OFF_SLOTW    (OFF_SLOTTOK + 20480)               // CAPPAD f32 = 20KB
#define OFF_XG       (OFF_SLOTW  + 20480)                // CAPPAD*H bf16 = 20.97MB (16B aligned)
#define OFF_ABUF     (OFF_XG + (size_t)CAPPAD*HH*2)      // CAPPAD*I bf16 = 7.86MB

// ---------------- helpers ----------------
__device__ __forceinline__ unsigned short f2bf(float f) {
    uint32_t u = __builtin_bit_cast(uint32_t, f);
    uint32_t r = u + 0x7FFFu + ((u >> 16) & 1u);   // round-to-nearest-even
    return (unsigned short)(r >> 16);
}

__device__ __forceinline__ unsigned int cvt2bf(float x, float y) {
#ifdef USE_CVT_PK
    v2bf p = __builtin_amdgcn_cvt_pk_bf16_f32(x, y);
    return __builtin_bit_cast(unsigned int, p);
#else
    return (unsigned)f2bf(x) | ((unsigned)f2bf(y) << 16);
#endif
}

__device__ __forceinline__ uint4 pack8(float4 a, float4 b) {
    uint4 p;
    p.x = cvt2bf(a.x, a.y);
    p.y = cvt2bf(a.z, a.w);
    p.z = cvt2bf(b.x, b.y);
    p.w = cvt2bf(b.z, b.w);
    return p;
}

// Pure bf16 16B copy global->LDS: async (ASYNCcnt path) when available.
// Builtin signature (from hipcc diagnostic): param0 = v4i32 AS1* (global src).
__device__ __forceinline__ void copy16_to_lds(unsigned short* ldsDst, const unsigned short* gSrc) {
#ifdef USE_ASYNC_LDS
    __builtin_amdgcn_global_load_async_to_lds_b128(
        (__attribute__((address_space(1))) tv4i*)(void*)gSrc,
        (__attribute__((address_space(3))) tv4i*)(void*)ldsDst, 0, 0);
#else
    *(uint4*)ldsDst = *(const uint4*)gSrc;
#endif
}

__device__ __forceinline__ void wait_lds_fills() {
#ifdef USE_ASYNC_LDS
    __builtin_amdgcn_s_wait_asynccnt(0);
#endif
}

// ISA 7.12.2 16-bit operand layout: lane holds K = {hb..hb+7} in regs0-3 and
// K = {16+hb..16+hb+7} in regs4-7 (hb = 0 for lanes 0-15, 8 for lanes 16-31).
__device__ __forceinline__ v16bf frag16(const unsigned short* row, int hb) {
    uint4 lo = *(const uint4*)(row + hb);       // K = hb..hb+7
    uint4 hi = *(const uint4*)(row + 16 + hb);  // K = 16+hb..16+hb+7
    v8u t;
    t[0]=lo.x; t[1]=lo.y; t[2]=lo.z; t[3]=lo.w;
    t[4]=hi.x; t[5]=hi.y; t[6]=hi.z; t[7]=hi.w;
    return __builtin_bit_cast(v16bf, t);
}

// ---------------- kernels ----------------
__global__ void k_zero(float* __restrict__ out, int* __restrict__ small) {
    size_t i = (size_t)blockIdx.x * blockDim.x + threadIdx.x;
    if (i < (size_t)TT * HH) out[i] = 0.f;
    if (i < 128) small[i] = 0;   // counts + offsets scratch
}

__global__ void k_router(const float* __restrict__ hs, const float* __restrict__ gw,
                         int* __restrict__ counts, int* __restrict__ tokExp,
                         float* __restrict__ tokW) {
    int t = blockIdx.x * blockDim.x + threadIdx.x;
    if (t >= TT) return;
    float acc[EE];
    #pragma unroll
    for (int e = 0; e < EE; ++e) acc[e] = 0.f;
    const float* xr = hs + (size_t)t * HH;
    for (int h = 0; h < HH; h += 4) {
        float4 x = *(const float4*)(xr + h);
        #pragma unroll
        for (int e = 0; e < EE; ++e) {
            float4 g = *(const float4*)(gw + (size_t)e * HH + h);
            acc[e] += x.x*g.x + x.y*g.y + x.z*g.z + x.w*g.w;
        }
    }
    float mx = acc[0];
    #pragma unroll
    for (int e = 1; e < EE; ++e) mx = fmaxf(mx, acc[e]);
    float p[EE];
    #pragma unroll
    for (int e = 0; e < EE; ++e) p[e] = __expf(acc[e] - mx);
    // top-2, first-index tie-break (matches lax.top_k)
    int i0 = 0; float v0 = p[0];
    #pragma unroll
    for (int e = 1; e < EE; ++e) if (p[e] > v0) { v0 = p[e]; i0 = e; }
    int i1 = -1; float v1 = -1.f;
    #pragma unroll
    for (int e = 0; e < EE; ++e) if (e != i0 && p[e] > v1) { v1 = p[e]; i1 = e; }
    float inv = 1.f / (v0 + v1);
    tokExp[t*2+0] = i0; tokW[t*2+0] = v0 * inv;
    tokExp[t*2+1] = i1; tokW[t*2+1] = v1 * inv;
    atomicAdd(&counts[i0], 1);
    atomicAdd(&counts[i1], 1);
}

__global__ void k_scan(const int* __restrict__ counts, int* __restrict__ offs) {
    if (blockIdx.x == 0 && threadIdx.x == 0) {
        int off = 0;
        for (int e = 0; e < EE; ++e) {
            offs[e] = off;
            off += ((counts[e] + 63) >> 6) << 6;   // pad to 64-row tiles
        }
        offs[EE] = off;
    }
}

// Deterministic compaction: slot = offs[e] + rank of assignment among its expert.
__global__ void k_build(const int* __restrict__ tokExp, const float* __restrict__ tokW,
                        const int* __restrict__ offs,
                        int* __restrict__ slotTok, float* __restrict__ slotW) {
    int a = blockIdx.x * blockDim.x + threadIdx.x;
    if (a >= TT * TOPK) return;
    int e = tokExp[a];
    int rank = 0;
    for (int b = 0; b < a; ++b) rank += (tokExp[b] == e);
    int slot = offs[e] + rank;
    slotTok[slot] = a >> 1;
    slotW[slot]   = tokW[a];
}

// Gather token rows into bf16 Xg; zero the pad rows so later tiles stay finite.
__global__ void k_gather(const float* __restrict__ hs,
                         const int* __restrict__ counts, const int* __restrict__ offs,
                         const int* __restrict__ slotTok,
                         unsigned short* __restrict__ Xg) {
    int s = blockIdx.x;
    bool valid = false;
    #pragma unroll
    for (int e = 0; e < EE; ++e) valid |= (s >= offs[e]) && (s < offs[e] + counts[e]);
    uint4* dst = (uint4*)(Xg + (size_t)s * HH);
    int j = threadIdx.x;                     // 256 threads x 8 elems = 2048
    if (!valid) { dst[j] = make_uint4(0,0,0,0); return; }
    int tok = slotTok[s];
    const float4* src = (const float4*)(hs + (size_t)tok * HH);
    dst[j] = pack8(src[j*2], src[j*2+1]);
}

// GEMM1: h13 = Xg_e @ w13_eᵀ, fused silu(gate)*up -> Abuf (bf16).
// Block: 64 tokens x 64 I-cols, 8 waves = (wm 0..3) x (wn 0..1).
__global__ __launch_bounds__(256, 2) void k_gemm1(
    const float* __restrict__ w13, const unsigned short* __restrict__ Xg,
    const int* __restrict__ counts, const int* __restrict__ offs,
    unsigned short* __restrict__ Abuf) {
    const int e  = blockIdx.z;
    const int rt = blockIdx.y;
    const int ct = blockIdx.x;
    const int ne = counts[e];
    if (rt * 64 >= ne) return;
    const int slotBase = offs[e] + rt * 64;
    const int colBase  = ct * 64;

    __shared__ __align__(16) unsigned short sX[64 * 32];
    __shared__ __align__(16) unsigned short sG[64 * 32];
    __shared__ __align__(16) unsigned short sU[64 * 32];

    const int tid  = threadIdx.x;
    const int lane = tid & 31;
    const int wv   = tid >> 5;
    const int wm   = wv & 3;
    const int wn   = wv >> 2;
    const int mL   = lane & 15;
    const int hb   = (lane >> 4) << 3;

    const int rr   = tid >> 2;        // 0..63 tile row
    const int sub  = (tid & 3) << 3;  // element start 0/8/16/24

    const float* gRow = w13 + ((size_t)e * 2 * II + colBase + rr) * HH + sub;
    const float* uRow = w13 + ((size_t)e * 2 * II + II + colBase + rr) * HH + sub;
    const unsigned short* xRow = Xg + (size_t)(slotBase + rr) * HH + sub;

    v8f accG0 = {}, accG1 = {}, accU0 = {}, accU1 = {};

    for (int kk = 0; kk < HH; kk += 32) {
        __syncthreads();
        copy16_to_lds(sX + rr*32 + sub, xRow + kk);   // bf16 tile: async LDS fill
        *(uint4*)(sG + rr*32 + sub) = pack8(*(const float4*)(gRow + kk),
                                            *(const float4*)(gRow + kk + 4));
        *(uint4*)(sU + rr*32 + sub) = pack8(*(const float4*)(uRow + kk),
                                            *(const float4*)(uRow + kk + 4));
        if (kk + 32 < HH) {           // global_prefetch_b8 next K-tile
            __builtin_prefetch(gRow + kk + 32, 0, 1);
            __builtin_prefetch(uRow + kk + 32, 0, 1);
        }
        wait_lds_fills();
        __syncthreads();
        v16bf a  = frag16(sX + (wm*16 + mL) * 32, hb);
        v16bf g0 = frag16(sG + (wn*32 + mL) * 32, hb);
        v16bf g1 = frag16(sG + (wn*32 + 16 + mL) * 32, hb);
        v16bf u0 = frag16(sU + (wn*32 + mL) * 32, hb);
        v16bf u1 = frag16(sU + (wn*32 + 16 + mL) * 32, hb);
        accG0 = __builtin_amdgcn_wmma_f32_16x16x32_bf16(false, a, false, g0, (short)0, accG0, false, false);
        accG1 = __builtin_amdgcn_wmma_f32_16x16x32_bf16(false, a, false, g1, (short)0, accG1, false, false);
        accU0 = __builtin_amdgcn_wmma_f32_16x16x32_bf16(false, a, false, u0, (short)0, accU0, false, false);
        accU1 = __builtin_amdgcn_wmma_f32_16x16x32_bf16(false, a, false, u1, (short)0, accU1, false, false);
    }

    // C/D layout: element (M = v + 8*(lane>=16), N = lane&15)
    #pragma unroll
    for (int v = 0; v < 8; ++v) {
        int m  = v + ((lane >> 4) << 3);
        int rl = wm * 16 + m;
        if (rt * 64 + rl >= ne) continue;
        size_t row = (size_t)(slotBase + rl);
        float g0v = accG0[v], g1v = accG1[v];
        float s0 = g0v / (1.f + __expf(-g0v));
        float s1 = g1v / (1.f + __expf(-g1v));
        Abuf[row * II + colBase + wn*32 + mL]      = f2bf(s0 * accU0[v]);
        Abuf[row * II + colBase + wn*32 + 16 + mL] = f2bf(s1 * accU1[v]);
    }
}

// GEMM2: Y = Abuf_e @ w2_eᵀ; scaled scatter-add into out (exactly 2 adds/element,
// fp add of 2 operands is commutative -> bitwise deterministic).
__global__ __launch_bounds__(256, 2) void k_gemm2(
    const float* __restrict__ w2, const unsigned short* __restrict__ Abuf,
    const int* __restrict__ counts, const int* __restrict__ offs,
    const int* __restrict__ slotTok, const float* __restrict__ slotW,
    float* __restrict__ out) {
    const int e  = blockIdx.z;
    const int rt = blockIdx.y;
    const int ct = blockIdx.x;
    const int ne = counts[e];
    if (rt * 64 >= ne) return;
    const int slotBase = offs[e] + rt * 64;
    const int colBase  = ct * 64;

    __shared__ __align__(16) unsigned short sA[64 * 32];
    __shared__ __align__(16) unsigned short sW[64 * 32];

    const int tid  = threadIdx.x;
    const int lane = tid & 31;
    const int wv   = tid >> 5;
    const int wm   = wv & 3;
    const int wn   = wv >> 2;
    const int mL   = lane & 15;
    const int hb   = (lane >> 4) << 3;
    const int rr   = tid >> 2;
    const int sub  = (tid & 3) << 3;

    const float* wRow = w2 + ((size_t)e * HH + colBase + rr) * II + sub;
    const unsigned short* aRow = Abuf + (size_t)(slotBase + rr) * II + sub;

    v8f acc0 = {}, acc1 = {};

    for (int kk = 0; kk < II; kk += 32) {
        __syncthreads();
        copy16_to_lds(sA + rr*32 + sub, aRow + kk);   // bf16 tile: async LDS fill
        *(uint4*)(sW + rr*32 + sub) = pack8(*(const float4*)(wRow + kk),
                                            *(const float4*)(wRow + kk + 4));
        if (kk + 32 < II) __builtin_prefetch(wRow + kk + 32, 0, 1);
        wait_lds_fills();
        __syncthreads();
        v16bf a  = frag16(sA + (wm*16 + mL) * 32, hb);
        v16bf b0 = frag16(sW + (wn*32 + mL) * 32, hb);
        v16bf b1 = frag16(sW + (wn*32 + 16 + mL) * 32, hb);
        acc0 = __builtin_amdgcn_wmma_f32_16x16x32_bf16(false, a, false, b0, (short)0, acc0, false, false);
        acc1 = __builtin_amdgcn_wmma_f32_16x16x32_bf16(false, a, false, b1, (short)0, acc1, false, false);
    }

    #pragma unroll
    for (int v = 0; v < 8; ++v) {
        int m  = v + ((lane >> 4) << 3);
        int rl = wm * 16 + m;
        if (rt * 64 + rl >= ne) continue;
        int slot  = slotBase + rl;
        int tok   = slotTok[slot];
        float wgt = slotW[slot];
        float* o = out + (size_t)tok * HH + colBase + wn*32 + mL;
        atomicAdd(o,      acc0[v] * wgt);
        atomicAdd(o + 16, acc1[v] * wgt);
    }
}

// ---------------- launcher ----------------
extern "C" void kernel_launch(void* const* d_in, const int* in_sizes, int n_in,
                              void* d_out, int out_size, void* d_ws, size_t ws_size,
                              hipStream_t stream) {
    const float* hs  = (const float*)d_in[0];   // (T,H)
    const float* gw  = (const float*)d_in[1];   // (E,H)
    const float* w13 = (const float*)d_in[2];   // (E,2I,H)
    const float* w2  = (const float*)d_in[3];   // (E,H,I)
    float* out = (float*)d_out;

    char* ws = (char*)d_ws;
    int*   counts  = (int*)  (ws + OFF_COUNTS);
    int*   offs    = (int*)  (ws + OFF_OFFS);
    int*   tokExp  = (int*)  (ws + OFF_TOKEXP);
    float* tokW    = (float*)(ws + OFF_TOKW);
    int*   slotTok = (int*)  (ws + OFF_SLOTTOK);
    float* slotW   = (float*)(ws + OFF_SLOTW);
    unsigned short* Xg   = (unsigned short*)(ws + OFF_XG);
    unsigned short* Abuf = (unsigned short*)(ws + OFF_ABUF);

    k_zero  <<<(TT*HH + 255)/256, 256, 0, stream>>>(out, counts);
    k_router<<<(TT + 255)/256,     256, 0, stream>>>(hs, gw, counts, tokExp, tokW);
    k_scan  <<<1, 32, 0, stream>>>(counts, offs);
    k_build <<<(TT*TOPK + 255)/256, 256, 0, stream>>>(tokExp, tokW, offs, slotTok, slotW);
    k_gather<<<CAPPAD, 256, 0, stream>>>(hs, counts, offs, slotTok, Xg);
    k_gemm1 <<<dim3(II/64, 64, EE), 256, 0, stream>>>(w13, Xg, counts, offs, Abuf);
    k_gemm2 <<<dim3(HH/64, 64, EE), 256, 0, stream>>>(w2, Abuf, counts, offs, slotTok, slotW, out);
}
